// NCAHarmonicController_79731772883062
// MI455X (gfx1250) — matrix operand order repaired and checked by reading, not compiled, not run on hardware
//
#include <hip/hip_runtime.h>
#include <hip/hip_bf16.h>
#include <math.h>

// CDNA5 / gfx1250: wave32, WMMA 16x16x32 f16 -> f32 accumulate.
typedef __attribute__((ext_vector_type(16))) _Float16 v16h;
typedef __attribute__((ext_vector_type(8)))  _Float16 v8h;
typedef __attribute__((ext_vector_type(2)))  _Float16 v2h;
typedef __attribute__((ext_vector_type(8)))  float    v8f;

static constexpr int kC   = 18;   // channels
static constexpr int kN   = 32;   // harmonics (positions)
static constexpr int kC3  = 54;   // 3*C perception width (padded to 64: slot 54 = bias-1)
static constexpr int kH1  = 64;
static constexpr int kH2  = 32;
static constexpr int kWaves = 4;  // waves per block (128 threads)
static constexpr float kAliveThresh = 0.1f;

__device__ __forceinline__ v8f wmma_f16(v16h a, v16h b, v8f c) {
  // (neg_a, A, neg_b, B, c_mod, C, reuse_a, reuse_b)
  return __builtin_amdgcn_wmma_f32_16x16x32_f16(false, a, false, b, (short)0, c,
                                                false, false);
}

// Pairwise f32->f16 convert (v_cvt_pk_f16_f32). Result is canonical, so the
// following f16 max needs no fcanonicalize. ReLU commutes with RTE truncation.
__device__ __forceinline__ v2h cvt2(float a, float b) {
  v2h h; h[0] = (_Float16)a; h[1] = (_Float16)b;
  return h;
}
__device__ __forceinline__ v2h relu2(v2h h) {
  const v2h z = {(_Float16)0.0f, (_Float16)0.0f};
  return __builtin_elementwise_max(h, z);   // v_pk_max_num_f16
}

// f16 B-operand (K=32 x N=16 tile) from row-major f32 weights.
// B layout (wave32): lane l holds column n = l&15; lane group g=l>>4: halves 0..7 hold
// K = kchunk*32 + g*8 + h, halves 8..15 hold K = kchunk*32 + 16 + g*8 + (h-8).
__device__ __forceinline__ v16h load_B_operand(const float* __restrict__ W,
                                               int Kreal, int Nreal,
                                               int ntile, int kchunk, int lane) {
  const int n     = ntile * 16 + (lane & 15);
  const int kbase = kchunk * 32 + ((lane >> 4) * 8);
  v16h r;
#pragma unroll
  for (int h = 0; h < 16; ++h) {
    const int K = kbase + h + (h >= 8 ? 8 : 0);
    float v = (K < Kreal && n < Nreal) ? W[K * Nreal + n] : 0.0f;
    r[h] = (_Float16)v;
  }
  return r;
}

// W1 B-operand with bias folded in: virtual row K=54 holds b1 (A supplies 1.0 there).
__device__ __forceinline__ v16h load_B1_operand(const float* __restrict__ W1,
                                                const float* __restrict__ b1,
                                                int ntile, int kchunk, int lane) {
  const int n     = ntile * 16 + (lane & 15);
  const int kbase = kchunk * 32 + ((lane >> 4) * 8);
  v16h r;
#pragma unroll
  for (int h = 0; h < 16; ++h) {
    const int K = kbase + h + (h >= 8 ? 8 : 0);
    float v = (K < kC3) ? W1[K * kH1 + n] : ((K == kC3) ? b1[n] : 0.0f);
    r[h] = (_Float16)v;
  }
  return r;
}

// f16 A-operand (M=16 x K=32) from an LDS row of halves (row = this lane's M).
// ISA A layout: lane group g=l>>4: halves 0..7 -> K = kc*32 + g*8 + h,
// halves 8..15 -> K = kc*32 + 16 + g*8 + (h-8). Both slices: 16B-aligned b128.
__device__ __forceinline__ v16h load_A_operand(const _Float16* __restrict__ row,
                                               int kchunk, int lane) {
  const int kb = kchunk * 32 + ((lane >> 4) * 8);
  v8h lo = *(const v8h*)(row + kb);
  v8h hi = *(const v8h*)(row + kb + 16);
  v16h a;
#pragma unroll
  for (int i = 0; i < 8; ++i) { a[i] = lo[i]; a[i + 8] = hi[i]; }
  return a;
}

__global__ __launch_bounds__(kWaves * 32)
void nca_step_kernel(const float* __restrict__ gin,
                     const float* __restrict__ W1, const float* __restrict__ b1,
                     const float* __restrict__ W2, const float* __restrict__ b2,
                     const float* __restrict__ W3, const float* __restrict__ b3,
                     float* __restrict__ gout, int Btot) {
  // Per-wave private LDS slices (no cross-wave sharing -> no barriers;
  // DS ops are in-order within a wave). H1/H2 double-buffered per tile so the
  // scheduler can interleave the two independent tile pipelines.
  __shared__ __align__(16) float    g_s [kWaves][kC * kN];      // staged grid row (f32)
  __shared__ __align__(16) float    U_s [kWaves][kC * kN];      // MLP update (f32)
  __shared__ __align__(16) _Float16 P_s [kWaves][kN * 64];      // perception rows
  __shared__ __align__(16) _Float16 H1_s[kWaves][2][16 * kH1];  // layer-1 act per tile
  __shared__ __align__(16) _Float16 H2_s[kWaves][2][16 * kH2];  // layer-2 act per tile
  __shared__ __align__(16) float    M_s [kWaves][kN];           // alive mask per cell

  const int lane = threadIdx.x & 31;
  const int wave = threadIdx.x >> 5;
  const int col  = lane & 15;           // this lane's C/D-layout column
  const int moff = (lane >> 4) * 8;     // this lane's C/D-layout M offset
  float*    gs = g_s[wave];
  float*    Us = U_s[wave];
  _Float16* Ps = P_s[wave];
  float*    Ms = M_s[wave];

  // ---- Resident f16 weight operands (built once; ~112 VGPRs) ----
  v16h B1w[4][2], B2w[2][2], B3w[2];
#pragma unroll
  for (int nt = 0; nt < 4; ++nt) {
    B1w[nt][0] = load_B1_operand(W1, b1, nt, 0, lane);
    B1w[nt][1] = load_B1_operand(W1, b1, nt, 1, lane);
  }
#pragma unroll
  for (int nt = 0; nt < 2; ++nt) {
    B2w[nt][0] = load_B_operand(W2, kH1, kH2, nt, 0, lane);
    B2w[nt][1] = load_B_operand(W2, kH1, kH2, nt, 1, lane);
    B3w[nt]    = load_B_operand(W3, kH2, kC,  nt, 0, lane);
  }
  // Layer-2 bias as packed f16 (both halves identical: pair shares a column);
  // layer-3 bias in f32 C-layout. b1 is folded into B1w.
  v2h   b2h[2];
  float b3v[2];
#pragma unroll
  for (int nt = 0; nt < 2; ++nt) {
    const _Float16 bh = (_Float16)b2[nt * 16 + col];
    b2h[nt][0] = bh; b2h[nt][1] = bh;
    const int cch = nt * 16 + col;
    b3v[nt] = (cch < kC) ? b3[cch] : 0.0f;
  }

  const int gw = blockIdx.x * kWaves + wave;
  const int nw = gridDim.x * kWaves;
  const int row_f4 = (kC * kN) / 4;  // 144 float4 per batch row

  for (int b = gw; b < Btot; b += nw) {
    // Prefetch next batch row while we work on this one.
    const int bn = b + nw;
    if (bn < Btot) __builtin_prefetch(gin + (size_t)bn * (kC * kN), 0, 1);

    // ---- Stage grid row: coalesced b128 global loads -> LDS ----
    const float4* gr4 = (const float4*)(gin + (size_t)b * (kC * kN));
#pragma unroll
    for (int i = 0; i < 5; ++i) {
      const int idx = lane + i * 32;
      if (idx < row_f4) *(float4*)&gs[idx * 4] = gr4[idx];
    }

    // ---- Build perception rows (cell n = lane): [left | center | right | 1 | 0..] ----
    {
      const int nm1 = (lane + 31) & 31, np1 = (lane + 1) & 31;
      _Float16* prow = Ps + lane * 64;
#pragma unroll
      for (int c = 0; c < kC; ++c) {
        prow[c]          = (_Float16)gs[c * kN + nm1];
        prow[kC + c]     = (_Float16)gs[c * kN + lane];
        prow[2 * kC + c] = (_Float16)gs[c * kN + np1];
      }
      prow[kC3] = (_Float16)1.0f;          // bias slot (pairs with b1 row in B1w)
#pragma unroll
      for (int k = kC3 + 1; k < 64; ++k) prow[k] = (_Float16)0.0f;
    }

    // ---- Two independent M=16 tiles cover the 32 cells of this batch row ----
#pragma unroll
    for (int t = 0; t < 2; ++t) {
      _Float16* H1 = H1_s[wave][t];
      _Float16* H2 = H2_s[wave][t];

      // Layer 1: 16x64 (K incl. bias) x W1 -> 16x64; ReLU in f16 after pk-convert.
      v16h a0 = load_A_operand(Ps + (t * 16 + col) * 64, 0, lane);
      v16h a1 = load_A_operand(Ps + (t * 16 + col) * 64, 1, lane);
#pragma unroll
      for (int nt = 0; nt < 4; ++nt) {
        v8f acc = wmma_f16(a1, B1w[nt][1], wmma_f16(a0, B1w[nt][0], v8f{}));
#pragma unroll
        for (int j = 0; j < 8; j += 2) {
          v2h h = relu2(cvt2(acc[j], acc[j + 1]));
          H1[(j + 0 + moff) * kH1 + nt * 16 + col] = h[0];
          H1[(j + 1 + moff) * kH1 + nt * 16 + col] = h[1];
        }
      }

      // Layer 2: 16x64 x W2 -> 16x32; bias + ReLU as packed f16.
      v16h h0 = load_A_operand(H1 + col * kH1, 0, lane);
      v16h h1 = load_A_operand(H1 + col * kH1, 1, lane);
#pragma unroll
      for (int nt = 0; nt < 2; ++nt) {
        v8f acc = wmma_f16(h1, B2w[nt][1], wmma_f16(h0, B2w[nt][0], v8f{}));
#pragma unroll
        for (int j = 0; j < 8; j += 2) {
          v2h h = relu2(cvt2(acc[j], acc[j + 1]) + b2h[nt]);
          H2[(j + 0 + moff) * kH2 + nt * 16 + col] = h[0];
          H2[(j + 1 + moff) * kH2 + nt * 16 + col] = h[1];
        }
      }

      // Layer 3: 16x32 x W3 -> 16x18 (N padded to 32). Plain stores into U_s.
      v16h h2 = load_A_operand(H2 + col * kH2, 0, lane);
#pragma unroll
      for (int nt = 0; nt < 2; ++nt) {
        v8f acc = wmma_f16(h2, B3w[nt], v8f{});
        const int cch = nt * 16 + col;
        if (cch < kC) {
#pragma unroll
          for (int j = 0; j < 8; ++j) {
            const int n = t * 16 + j + moff;        // cell position
            Us[cch * kN + n] = acc[j] + b3v[nt];    // unique (c,n) per lane
          }
        }
      }
    } // t

    // ---- Epilogue: sigmoid on channel 1, alive mask, fused (g+u)*mask store ----
    {
      const float pre1  = gs[1 * kN + lane] + Us[1 * kN + lane];
      const float alpha = 1.0f / (1.0f + __expf(-pre1));
      gs[1 * kN + lane] = alpha;   // channel-1 output value (pre-mask)
      Us[1 * kN + lane] = 0.0f;    // so the fused add below is a no-op for ch.1
      Ms[lane]          = (alpha > kAliveThresh) ? 1.0f : 0.0f;
    }
    float4* o4 = (float4*)(gout + (size_t)b * (kC * kN));
#pragma unroll
    for (int i = 0; i < 5; ++i) {
      const int idx = lane + i * 32;
      if (idx < row_f4) {
        float4 g = *(const float4*)&gs[idx * 4];
        float4 u = *(const float4*)&Us[idx * 4];
        const int n0 = (idx * 4) & (kN - 1);  // a float4 never crosses a channel row
        float4 v;
        v.x = (g.x + u.x) * Ms[n0 + 0];
        v.y = (g.y + u.y) * Ms[n0 + 1];
        v.z = (g.z + u.z) * Ms[n0 + 2];
        v.w = (g.w + u.w) * Ms[n0 + 3];
        o4[idx] = v;
      }
    }
  }
}

extern "C" void kernel_launch(void* const* d_in, const int* in_sizes, int n_in,
                              void* d_out, int out_size, void* d_ws, size_t ws_size,
                              hipStream_t stream) {
  const float* gin = (const float*)d_in[0];
  const float* W1  = (const float*)d_in[1];
  const float* b1  = (const float*)d_in[2];
  const float* W2  = (const float*)d_in[3];
  const float* b2  = (const float*)d_in[4];
  const float* W3  = (const float*)d_in[5];
  const float* b3  = (const float*)d_in[6];
  float* out = (float*)d_out;

  const int Btot = in_sizes[0] / (kC * kN);   // 65536
  dim3 grd(4096);                             // 16384 waves, 4 batch rows each
  dim3 blk(kWaves * 32);
  hipLaunchKernelGGL(nca_step_kernel, grd, blk, 0, stream,
                     gin, W1, b1, W2, b2, W3, b3, out, Btot);
}
